// GNN_69088843923647
// MI455X (gfx1250) — matrix operand order repaired and checked by reading, not compiled, not run on hardware
//
#include <hip/hip_runtime.h>
#include <hip/hip_bf16.h>
#include <stdint.h>
#include <stddef.h>

// ---------------------------------------------------------------------------
// GraphSAGE 3-layer forward for MI455X (gfx1250, wave32, WMMA)
// - Node features kept in bf16 (halves random gather traffic on 23.3TB/s HBM)
// - Aggregation accumulates f32 via global_atomic_add_f32
// - Dense matmuls: v_wmma_f32_16x16x32_bf16, one wave per 16xN row-block so
//   A fragments are loaded once and reused across all N-tiles (8x less A
//   traffic than tile-per-wave); packed weight fragments stay L2-resident.
// ---------------------------------------------------------------------------

typedef __attribute__((ext_vector_type(16))) __bf16 v16bf;
typedef __attribute__((ext_vector_type(8)))  __bf16 v8bf;
typedef __attribute__((ext_vector_type(4)))  __bf16 v4bf;
typedef __attribute__((ext_vector_type(8)))  float  v8f;

#define FEAT 128            // K dimension (IN_F == H_F == 128)
#define KT   4              // 128 / 32 k-tiles per wmma chain
#define BN_EPS 1e-5f

static __device__ __forceinline__ v16bf cat8(v8bf lo, v8bf hi) {
    v16bf r;
#pragma unroll
    for (int i = 0; i < 8; ++i) { r[i] = lo[i]; r[i + 8] = hi[i]; }
    return r;
}

static __device__ __forceinline__ uint32_t hash_u32(uint32_t x) {
    x ^= x >> 16; x *= 0x7feb352du;
    x ^= x >> 15; x *= 0x846ca68bu;
    x ^= x >> 16; return x;
}

// ---------------- elementwise helpers ----------------

__global__ void k_fill0(float* p, size_t n) {
    size_t i = (size_t)blockIdx.x * blockDim.x + threadIdx.x;
    size_t stride = (size_t)gridDim.x * blockDim.x;
    for (; i < n; i += stride) p[i] = 0.0f;
}

__global__ void k_f32_to_bf16(const float* __restrict__ in, __bf16* __restrict__ out, size_t n) {
    size_t i = (size_t)blockIdx.x * blockDim.x + threadIdx.x;
    size_t stride = (size_t)gridDim.x * blockDim.x;
    for (; i < n; i += stride) out[i] = (__bf16)in[i];
}

// Pack a row-major KxN f32 weight into per-(ntile,ktile) WMMA B fragments:
// frag index = nt*KT + kt ; within a frag: lane*16 + i, where the lane holds
// column n = nt*16 + (lane&15) and K = kt*32 + (i<8 ? i : i+8) + (lane>>4)*8  (i = 0..15).
__global__ void k_pack_w(const float* __restrict__ W, __bf16* __restrict__ out, int K, int N) {
    int NT = N >> 4;
    int total = (K >> 5) * NT * 512;
    int t = blockIdx.x * blockDim.x + threadIdx.x;
    if (t >= total) return;
    int frag = t >> 9;
    int r    = t & 511;
    int lane = r >> 4;
    int i    = r & 15;
    int nt = frag / KT;
    int kt = frag % KT;
    int n = nt * 16 + (lane & 15);
    int k = kt * 32 + ((lane >> 4) * 8) + (i < 8 ? i : i + 8);
    out[t] = (__bf16)W[(size_t)k * N + n];
}

// ---------------- edge aggregation ----------------
// 32 threads per edge; each thread handles 4 features (8B bf16 load, 4 f32 atomics).
__global__ void k_agg(const __bf16* __restrict__ h, const int* __restrict__ src,
                      const int* __restrict__ dst, float* __restrict__ msg,
                      float* __restrict__ deg, int nEdges) {
    long t = (long)blockIdx.x * blockDim.x + threadIdx.x;
    int e = (int)(t >> 5);
    int lane = (int)(t & 31);
    if (e >= nEdges) return;
    int s = src[e];
    int d = dst[e];
    const v4bf hv = *(const v4bf*)(h + (size_t)s * FEAT + lane * 4);
    float* mp = msg + (size_t)d * FEAT + lane * 4;
    atomicAdd(mp + 0, (float)hv[0]);
    atomicAdd(mp + 1, (float)hv[1]);
    atomicAdd(mp + 2, (float)hv[2]);
    atomicAdd(mp + 3, (float)hv[3]);
    if (lane == 0) atomicAdd(deg + d, 1.0f);
}

// msg_bf16 = bf16(msg / max(deg,1))
__global__ void k_norm(const float* __restrict__ msg, const float* __restrict__ deg,
                       __bf16* __restrict__ out, size_t n) {
    size_t i = (size_t)blockIdx.x * blockDim.x + threadIdx.x;
    size_t stride = (size_t)gridDim.x * blockDim.x;
    for (; i < n; i += stride) {
        float dg = deg[i >> 7];
        dg = dg > 1.0f ? dg : 1.0f;
        out[i] = (__bf16)(msg[i] / dg);
    }
}

// ---------------- fused SAGE GEMM (WMMA bf16 -> f32) ----------------
// out[M x N] = A_self[M x 128] @ Wself + A_neigh[M x 128] @ Wneigh + bias
// One wave per 16-row block: A fragments (both matrices, all k-tiles) are
// loaded once into 64 VGPRs and reused across every N-tile; only the tiny
// packed B fragments (L2-resident) are re-fetched per N-tile.
__global__ void __launch_bounds__(128)
k_sage_gemm(const __bf16* __restrict__ A0, const __bf16* __restrict__ A1,
            const __bf16* __restrict__ W0p, const __bf16* __restrict__ W1p,
            const float* __restrict__ bias, float* __restrict__ out,
            int M, int N) {
    const int lane = threadIdx.x & 31;
    const int mt   = blockIdx.x * (blockDim.x >> 5) + (threadIdx.x >> 5);
    const int Ntiles = N >> 4;
    const int Mtiles = M >> 4;
    if (mt >= Mtiles) return;              // wave-uniform: EXEC stays all-1s

    // ---- load all A fragments for this 16-row block (once) ----
    // A fragment: lane row = lane&15; element i -> K = (i<8 ? i : i+8) + (lane>=16 ? 8 : 0)
    const int row = mt * 16 + (lane & 15);
    const int kb  = (lane >> 4) * 8;
    const __bf16* ar0 = A0 + (size_t)row * FEAT;
    const __bf16* ar1 = A1 + (size_t)row * FEAT;
    __builtin_prefetch(ar0, 0, 0);         // global_prefetch_b8
    __builtin_prefetch(ar1, 0, 0);

    v16bf a0[KT], a1[KT];
#pragma unroll
    for (int kt = 0; kt < KT; ++kt) {
        const int k0 = kt * 32 + kb;
        a0[kt] = cat8(*(const v8bf*)(ar0 + k0), *(const v8bf*)(ar0 + k0 + 16));
        a1[kt] = cat8(*(const v8bf*)(ar1 + k0), *(const v8bf*)(ar1 + k0 + 16));
    }

    // ---- sweep N-tiles, reloading only the packed weight fragments ----
    const int cl = lane & 15;              // output column within tile / bias lane
    const int r0 = mt * 16 + ((lane >> 4) * 8);
    for (int nt = 0; nt < Ntiles; ++nt) {
        const float bv = bias[nt * 16 + cl];
        v8f acc;
#pragma unroll
        for (int i = 0; i < 8; ++i) acc[i] = bv;

#pragma unroll
        for (int kt = 0; kt < KT; ++kt) {
            const __bf16* p0 = W0p + ((size_t)(nt * KT + kt) << 9) + lane * 16;
            const __bf16* p1 = W1p + ((size_t)(nt * KT + kt) << 9) + lane * 16;
            v16bf bs = cat8(*(const v8bf*)p0, *(const v8bf*)(p0 + 8));
            acc = __builtin_amdgcn_wmma_f32_16x16x32_bf16(false, a0[kt], false, bs,
                                                          (short)0, acc, false, false);
            v16bf bn = cat8(*(const v8bf*)p1, *(const v8bf*)(p1 + 8));
            acc = __builtin_amdgcn_wmma_f32_16x16x32_bf16(false, a1[kt], false, bn,
                                                          (short)0, acc, false, false);
        }

        // C/D layout: VGPR i -> row (lane<16 ? i : 8+i), column = lane&15.
        const int c = nt * 16 + cl;
#pragma unroll
        for (int i = 0; i < 8; ++i)
            out[(size_t)(r0 + i) * N + c] = acc[i];
    }
}

// ---------------- BN statistics (column sums) ----------------
// block = 128 threads = 128 columns -> fully coalesced row reads.
__global__ void k_colstats(const float* __restrict__ pre, float* __restrict__ stats, int M) {
    const int col = threadIdx.x;
    float s = 0.0f, s2 = 0.0f;
    for (int r = blockIdx.x; r < M; r += gridDim.x) {
        float v = pre[(size_t)r * FEAT + col];
        s += v; s2 += v * v;
    }
    atomicAdd(&stats[col], s);
    atomicAdd(&stats[FEAT + col], s2);
}

// ---------------- fused BN(train) + ReLU + dropout -> bf16 ----------------
__global__ void k_bn_relu_drop(const float* __restrict__ pre, const float* __restrict__ stats,
                               const float* __restrict__ gamma, const float* __restrict__ beta,
                               __bf16* __restrict__ out, int M, uint32_t seed) {
    size_t n = (size_t)M * FEAT;
    size_t i = (size_t)blockIdx.x * blockDim.x + threadIdx.x;
    size_t stride = (size_t)gridDim.x * blockDim.x;
    const float invM = 1.0f / (float)M;
    for (; i < n; i += stride) {
        int col = (int)(i & (FEAT - 1));
        float mean = stats[col] * invM;
        float var  = stats[FEAT + col] * invM - mean * mean;
        float rstd = rsqrtf(var + BN_EPS);
        float v = (pre[i] - mean) * rstd * gamma[col] + beta[col];
        v = v > 0.0f ? v : 0.0f;
        uint32_t hsh = hash_u32((uint32_t)i * 2654435761u ^ seed);
        v = ((hsh & 0xFFFFu) >= 32768u) ? v * 2.0f : 0.0f;   // p = 0.5, scale 1/(1-p)
        out[i] = (__bf16)v;
    }
}

// ---------------------------------------------------------------------------
extern "C" void kernel_launch(void* const* d_in, const int* in_sizes, int n_in,
                              void* d_out, int out_size, void* d_ws, size_t ws_size,
                              hipStream_t stream) {
    const float* x   = (const float*)d_in[0];
    const int*   src = (const int*)d_in[1];
    const int*   dst = (const int*)d_in[2];
    const float* Ws0 = (const float*)d_in[3];
    const float* Wn0 = (const float*)d_in[4];
    const float* b0  = (const float*)d_in[5];
    const float* g0  = (const float*)d_in[6];
    const float* be0 = (const float*)d_in[7];
    const float* Ws1 = (const float*)d_in[8];
    const float* Wn1 = (const float*)d_in[9];
    const float* b1  = (const float*)d_in[10];
    const float* g1  = (const float*)d_in[11];
    const float* be1 = (const float*)d_in[12];
    const float* Ws2 = (const float*)d_in[13];
    const float* Wn2 = (const float*)d_in[14];
    const float* b2  = (const float*)d_in[15];

    const int M = in_sizes[0] / FEAT;      // 100000 (divisible by 16)
    const int E = in_sizes[1];             // 1600000
    const int NCLS = 64;
    const size_t MF = (size_t)M * FEAT;

    // ---- workspace carve-up (256B aligned regions) ----
    char* p = (char*)d_ws;
    auto carve = [&](size_t bytes) -> char* {
        char* r = p;
        p += (bytes + 255) & ~(size_t)255;
        return r;
    };
    float*  msg   = (float*) carve(MF * 4);
    float*  pre   = (float*) carve(MF * 4);
    float*  deg   = (float*) carve((size_t)M * 4);
    float*  stats = (float*) carve(256 * 4);
    __bf16* hA    = (__bf16*)carve(MF * 2);
    __bf16* hB    = (__bf16*)carve(MF * 2);
    __bf16* msgb  = (__bf16*)carve(MF * 2);
    __bf16* ws0p  = (__bf16*)carve(16384 * 2);   // 128x128: 4*8 frags * 512
    __bf16* wn0p  = (__bf16*)carve(16384 * 2);
    __bf16* ws1p  = (__bf16*)carve(16384 * 2);
    __bf16* wn1p  = (__bf16*)carve(16384 * 2);
    __bf16* ws2p  = (__bf16*)carve(8192 * 2);    // 128x64: 4*4 frags * 512
    __bf16* wn2p  = (__bf16*)carve(8192 * 2);

    const int TB = 256;
    const unsigned ew_blocks  = (unsigned)((MF + TB - 1) / TB);            // 50000
    const unsigned agg_blocks = (unsigned)(((size_t)E * 32 + TB - 1) / TB);
    const unsigned gemm_blocks = (unsigned)((M / 16 + 3) / 4);             // 4 waves/block

    // ---- one-time prep: x -> bf16, pack all weights ----
    k_f32_to_bf16<<<ew_blocks, TB, 0, stream>>>(x, hA, MF);
    k_pack_w<<<(16384 + TB - 1) / TB, TB, 0, stream>>>(Ws0, ws0p, FEAT, FEAT);
    k_pack_w<<<(16384 + TB - 1) / TB, TB, 0, stream>>>(Wn0, wn0p, FEAT, FEAT);
    k_pack_w<<<(16384 + TB - 1) / TB, TB, 0, stream>>>(Ws1, ws1p, FEAT, FEAT);
    k_pack_w<<<(16384 + TB - 1) / TB, TB, 0, stream>>>(Wn1, wn1p, FEAT, FEAT);
    k_pack_w<<<(8192 + TB - 1) / TB, TB, 0, stream>>>(Ws2, ws2p, FEAT, NCLS);
    k_pack_w<<<(8192 + TB - 1) / TB, TB, 0, stream>>>(Wn2, wn2p, FEAT, NCLS);

    // ---------------- layer 0 ----------------
    k_fill0<<<ew_blocks, TB, 0, stream>>>(msg, MF);
    k_fill0<<<(M + TB - 1) / TB, TB, 0, stream>>>(deg, (size_t)M);
    k_agg<<<agg_blocks, TB, 0, stream>>>(hA, src, dst, msg, deg, E);
    k_norm<<<ew_blocks, TB, 0, stream>>>(msg, deg, msgb, MF);
    k_sage_gemm<<<gemm_blocks, 128, 0, stream>>>(hA, msgb, ws0p, wn0p, b0, pre, M, FEAT);
    k_fill0<<<1, 256, 0, stream>>>(stats, 256);
    k_colstats<<<512, FEAT, 0, stream>>>(pre, stats, M);
    k_bn_relu_drop<<<ew_blocks, TB, 0, stream>>>(pre, stats, g0, be0, hB, M, 0x9E3779B9u);

    // ---------------- layer 1 ----------------
    k_fill0<<<ew_blocks, TB, 0, stream>>>(msg, MF);
    k_fill0<<<(M + TB - 1) / TB, TB, 0, stream>>>(deg, (size_t)M);
    k_agg<<<agg_blocks, TB, 0, stream>>>(hB, src, dst, msg, deg, E);
    k_norm<<<ew_blocks, TB, 0, stream>>>(msg, deg, msgb, MF);
    k_sage_gemm<<<gemm_blocks, 128, 0, stream>>>(hB, msgb, ws1p, wn1p, b1, pre, M, FEAT);
    k_fill0<<<1, 256, 0, stream>>>(stats, 256);
    k_colstats<<<512, FEAT, 0, stream>>>(pre, stats, M);
    k_bn_relu_drop<<<ew_blocks, TB, 0, stream>>>(pre, stats, g1, be1, hA, M, 0x85EBCA6Bu);

    // ---------------- layer 2 (no BN/ReLU/dropout) ----------------
    k_fill0<<<ew_blocks, TB, 0, stream>>>(msg, MF);
    k_fill0<<<(M + TB - 1) / TB, TB, 0, stream>>>(deg, (size_t)M);
    k_agg<<<agg_blocks, TB, 0, stream>>>(hA, src, dst, msg, deg, E);
    k_norm<<<ew_blocks, TB, 0, stream>>>(msg, deg, msgb, MF);
    k_sage_gemm<<<gemm_blocks, 128, 0, stream>>>(hA, msgb, ws2p, wn2p, b2,
                                                 (float*)d_out, M, NCLS);
}
